// dVAE_75041668596186
// MI455X (gfx1250) — compile-verified, matrix operand order checked by reading, not run on hardware
//
#include <hip/hip_runtime.h>
#include <hip/hip_bf16.h>
#include <math.h>

// ---------------------------------------------------------------------------
// Types for CDNA5 WMMA (wave32)
// ---------------------------------------------------------------------------
typedef __bf16 bf16_t;
typedef __bf16 v16bf __attribute__((ext_vector_type(16)));
typedef __bf16 v8bf  __attribute__((ext_vector_type(8)));
typedef float  v8f   __attribute__((ext_vector_type(8)));
typedef unsigned int u32x4 __attribute__((ext_vector_type(4)));
typedef int i32x4 __attribute__((ext_vector_type(4)));
typedef int i32x8 __attribute__((ext_vector_type(8)));

template <typename T> struct V8;
template <> struct V8<float>  { typedef float  type __attribute__((ext_vector_type(8))); };
template <> struct V8<__bf16> { typedef __bf16 type __attribute__((ext_vector_type(8))); };

#define BETA_CONST 6.6f
#define COUTP 128          // padded output-channel count of pre-transposed weights
#define KSTEP 64           // K elements staged per round (two 16x16x32 WMMA steps)

// Tensor Data Mover availability (device pass only; host pass -> fallback)
#if defined(__gfx1250__) && __has_builtin(__builtin_amdgcn_tensor_load_to_lds) && \
    __has_builtin(__builtin_amdgcn_s_wait_tensorcnt)
#define HAVE_TDM 1
#else
#define HAVE_TDM 0
#endif

// ---------------------------------------------------------------------------
// Weight prep: w (OIHW fp32, or [Cin,Cout,kh,kw] for ConvTranspose)
//   -> wT[o][k] bf16, o in [0,COUTP), k in [0,Kpad), zero padded.
// Row-major per o so the B tile is a contiguous 2-D tile (TDM friendly).
// ---------------------------------------------------------------------------
__global__ __launch_bounds__(256)
void prep_weight_kernel(const float* __restrict__ w, bf16_t* __restrict__ wT,
                        int Cout, int Cin, int KH, int KW, int Kpad, int transposed)
{
    int t = blockIdx.x * blockDim.x + threadIdx.x;
    if (t >= COUTP * Kpad) return;
    int o = t / Kpad, k = t - o * Kpad;
    int K = Cin * KH * KW;
    float v = 0.0f;
    if (o < Cout && k < K) {
        int ci = k / (KH * KW);
        int r2 = k % (KH * KW);
        int ky = r2 / KW, kx = r2 % KW;
        v = transposed ? w[((ci * Cout + o) * KH + ky) * KW + kx]
                       : w[((o * Cin + ci) * KH + ky) * KW + kx];
    }
    wT[(size_t)o * Kpad + k] = (bf16_t)v;
}

// ---------------------------------------------------------------------------
// Conv / transposed-conv implicit GEMM on v_wmma_f32_16x16x32_bf16.
// 4 waves (128 threads), tile (WM*64) x (WN*16); 4 M-subtiles per wave.
// Double-buffered LDS (WM<=2); B tile staged by the Tensor Data Mover when
// available (wave0 issues tensor_load_to_lds, waits s_wait_tensorcnt 0).
// ---------------------------------------------------------------------------
template <typename TIN, typename TOUT, int WM, int WN>
__global__ __launch_bounds__(128)
void conv_wmma(const TIN*  __restrict__ in,
               const bf16_t* __restrict__ wT,     // [COUTP][Kpad] bf16
               const float* __restrict__ bias,    // may be null
               const TOUT* __restrict__ resid,    // may be null
               TOUT* __restrict__ out,
               int Cin, int Hin, int Win, int lHinWin, int lWin,
               int Cout, int lHoW, int lWout, int Wout,
               int KH, int KW, int K, int Kpad, int sshift, int pad,
               int M, int in_relu, int out_relu, int transposed)
{
    constexpr int MI      = 4;
    constexpr int BLOCK_M = WM * MI * 16;
    constexpr int BLOCK_N = WN * 16;
    constexpr int THREADS = 128;
    constexpr int NBUF    = (WM <= 2) ? 2 : 1;   // LDS budget guard

    const int HoW  = 1 << lHoW;
    const int KHKW = KH * KW;
    const int Mtiles = M / BLOCK_M;              // M is a multiple of BLOCK_M
    const int mt = (int)blockIdx.x % Mtiles;
    const int nt = (int)blockIdx.x / Mtiles;
    const int mbase = mt * BLOCK_M;
    const int nbase = nt * BLOCK_N;

    __shared__ bf16_t Asm[NBUF][BLOCK_M][72];    // 144B row stride (16B chunks)
    __shared__ bf16_t Bsm[NBUF][BLOCK_N][72];

    const int tid  = threadIdx.x;
    const int lane = tid & 31;
    const int wid  = tid >> 5;
    const int wm   = wid % WM;
    const int wn   = wid / WM;
    const int r16  = lane & 15;
    const int lh   = lane >> 4;
    const int smask = (1 << sshift) - 1;

    v8f acc[MI];
#pragma unroll
    for (int mi = 0; mi < MI; ++mi) acc[mi] = (v8f){};

    // ---- A staging: 8 batched clamped loads -> one ds_store_b128 ----------
    auto stageA = [&](int kst, int bufi) {
        for (int m = tid; m < BLOCK_M; m += THREADS) {
            int gm  = mbase + m;
            int nb  = gm >> lHoW;
            int rem = gm & (HoW - 1);
            int oh  = rem >> lWout;
            int ow  = rem & (Wout - 1);
            int base = (nb * Cin) << lHinWin;
            int ohs, ows;
            if (!transposed) { ohs = (oh << sshift) - pad; ows = (ow << sshift) - pad; }
            else             { ohs = oh + pad;             ows = ow + pad; }
            int ci = kst / KHKW;
            int r2 = kst - ci * KHKW;
            int ky = r2 / KW;
            int kx = r2 - ky * KW;
            for (int k0 = 0; k0 < KSTEP; k0 += 8) {
                float tmp[8];
#pragma unroll
                for (int u = 0; u < 8; ++u) {
                    int gk = kst + k0 + u;
                    int ih, iw; bool ok;
                    if (!transposed) {
                        ih = ohs + ky; iw = ows + kx;
                        ok = ((unsigned)ih < (unsigned)Hin) & ((unsigned)iw < (unsigned)Win);
                    } else {
                        int tih = ohs - ky, tiw = ows - kx;
                        ok = (tih >= 0) & (tiw >= 0) &
                             ((tih & smask) == 0) & ((tiw & smask) == 0);
                        ih = tih >> sshift; iw = tiw >> sshift;
                        ok = ok & (ih < Hin) & (iw < Win);
                    }
                    ok = ok & (gk < K);
                    // clamped unconditional load (batchable), then select
                    int cic = ci < Cin ? ci : Cin - 1;
                    int ihc = ih < 0 ? 0 : (ih >= Hin ? Hin - 1 : ih);
                    int iwc = iw < 0 ? 0 : (iw >= Win ? Win - 1 : iw);
                    float lv = (float)in[base + (cic << lHinWin) + (ihc << lWin) + iwc];
                    tmp[u] = ok ? lv : 0.0f;
                    if (++kx == KW) { kx = 0; if (++ky == KH) { ky = 0; ++ci; } }
                }
                v8bf pk;
#pragma unroll
                for (int u = 0; u < 8; ++u) {
                    float v = tmp[u];
                    if (in_relu) v = fmaxf(v, 0.0f);
                    pk[u] = (bf16_t)v;
                }
                *(v8bf*)&Asm[bufi][m][k0] = pk;
            }
        }
    };

#if !HAVE_TDM
    // ---- B staging fallback: pure 16B copies (coalesced) -------------------
    auto stageB = [&](int kst, int bufi) {
        constexpr int KCH = KSTEP / 8;
        constexpr int CH  = BLOCK_N * KCH;
        for (int c = tid; c < CH; c += THREADS) {
            int n  = c / KCH;
            int kc = (c - n * KCH) * 8;
            v8bf pk = *(const v8bf*)(wT + (size_t)(nbase + n) * Kpad + kst + kc);
            *(v8bf*)&Bsm[bufi][n][kc] = pk;
        }
    };
#endif

    auto stageAll = [&](int kst, int bufi) {
#if HAVE_TDM
        if (wid == 0) {
            // Tensor DMA: 2-D tile BLOCK_N rows x KSTEP bf16, LDS row padded
            // 128B data + 16B pad (pad_interval=32 dwords, pad_amount=4 dwords)
            unsigned long long ga =
                (unsigned long long)(const void*)(wT + (size_t)nbase * Kpad + kst);
            unsigned ldsa = (unsigned)(unsigned long long)(const void*)&Bsm[bufi][0][0];
            u32x4 g0;
            g0[0] = 1u;                                   // count=1 (valid user D#)
            g0[1] = ldsa;                                 // lds_addr
            g0[2] = (unsigned)(ga & 0xFFFFFFFFu);         // global_addr[31:0]
            g0[3] = (unsigned)((ga >> 32) & 0x1FFFFFFu)   // global_addr[56:32]
                    | (2u << 30);                         // type=2 ("image")
            i32x8 g1;
            g1[0] = (int)((1u << 16)                      // data_size = 2B
                        | (1u << 20)                      // pad_enable
                        | (4u << 22)                      // pad_interval: 32 dwords
                        | (3u << 25));                    // pad_amount : 4 dwords
            unsigned td0 = (unsigned)(Kpad - kst);        // remaining dim0
            unsigned td1 = (unsigned)(COUTP - nbase);     // remaining dim1
            g1[1] = (int)((td0 & 0xFFFFu) << 16);
            g1[2] = (int)(((td0 >> 16) & 0xFFFFu) | ((td1 & 0xFFFFu) << 16));
            g1[3] = (int)(((td1 >> 16) & 0xFFFFu) | ((unsigned)KSTEP << 16)); // tile_dim0
            g1[4] = (int)((unsigned)BLOCK_N & 0xFFFFu);   // tile_dim1; tile_dim2=0
            unsigned long long st0 = (unsigned long long)Kpad;  // dim0 stride
            g1[5] = (int)(st0 & 0xFFFFFFFFu);
            g1[6] = (int)((st0 >> 32) & 0xFFFFu);
            g1[7] = 0;
            i32x4 g2 = {0, 0, 0, 0};
            i32x4 g3 = {0, 0, 0, 0};
#if __clang_major__ >= 23
            i32x8 g4 = {0, 0, 0, 0, 0, 0, 0, 0};
            __builtin_amdgcn_tensor_load_to_lds(g0, g1, g2, g3, g4, 0);
#else
            __builtin_amdgcn_tensor_load_to_lds(g0, g1, g2, g3, 0);
#endif
        }
        stageA(kst, bufi);
#else
        stageA(kst, bufi);
        stageB(kst, bufi);
#endif
    };

    auto waitStage = [&]() {
#if HAVE_TDM
        if (wid == 0) __builtin_amdgcn_s_wait_tensorcnt(0);
#endif
    };

    auto doWmma = [&](int bufi) {
#pragma unroll
        for (int ks = 0; ks < 2; ++ks) {
            v8bf blo = *(const v8bf*)&Bsm[bufi][wn * 16 + r16][ks * 32 + lh * 8];
            v8bf bhi = *(const v8bf*)&Bsm[bufi][wn * 16 + r16][ks * 32 + 16 + lh * 8];
            v16bf bfr;
#pragma unroll
            for (int q = 0; q < 8; ++q) { bfr[q] = blo[q]; bfr[8 + q] = bhi[q]; }
#pragma unroll
            for (int mi = 0; mi < MI; ++mi) {
                int am = wm * (MI * 16) + mi * 16 + r16;
                v8bf alo = *(const v8bf*)&Asm[bufi][am][ks * 32 + lh * 8];
                v8bf ahi = *(const v8bf*)&Asm[bufi][am][ks * 32 + 16 + lh * 8];
                v16bf afr;
#pragma unroll
                for (int q = 0; q < 8; ++q) { afr[q] = alo[q]; afr[8 + q] = ahi[q]; }
                acc[mi] = __builtin_amdgcn_wmma_f32_16x16x32_bf16(
                    false, afr, false, bfr, (short)0, acc[mi], false, false);
            }
        }
    };

    // ---- pipeline: prologue stage, then ping-pong ---------------------------
    stageAll(0, 0);
    waitStage();
    __syncthreads();
    int buf = 0;
    for (int kk = 0; kk < K; kk += KSTEP) {
        int nxt = kk + KSTEP;
        if constexpr (NBUF == 2) {
            if (nxt < K) stageAll(nxt, buf ^ 1);
            doWmma(buf);
            if (nxt < K) waitStage();
            __syncthreads();
            buf ^= 1;
        } else {
            doWmma(buf);
            __syncthreads();
            if (nxt < K) { stageAll(nxt, buf); waitStage(); __syncthreads(); }
        }
    }

    // ---- epilogue: 8 consecutive pixels per lane, vector store --------------
    typedef typename V8<TOUT>::type tv8;
    const int o = nbase + wn * 16 + r16;
    if (o < Cout) {
        float bval = bias ? bias[o] : 0.0f;
#pragma unroll
        for (int mi = 0; mi < MI; ++mi) {
            int gm0  = mbase + wm * (MI * 16) + mi * 16 + lh * 8;   // multiple of 8
            int nb   = gm0 >> lHoW;
            int rem0 = gm0 & (HoW - 1);
            int off0 = ((nb * Cout + o) << lHoW) + rem0;
            float vals[8];
#pragma unroll
            for (int r = 0; r < 8; ++r) vals[r] = acc[mi][r] + bval;
            if (resid) {
                tv8 rv = *(const tv8*)(resid + off0);
#pragma unroll
                for (int r = 0; r < 8; ++r) vals[r] += (float)rv[r];
            }
            if (out_relu) {
#pragma unroll
                for (int r = 0; r < 8; ++r) vals[r] = fmaxf(vals[r], 0.0f);
            }
            tv8 ov;
#pragma unroll
            for (int r = 0; r < 8; ++r) ov[r] = (TOUT)vals[r];
            *(tv8*)(out + off0) = ov;
        }
    }
}

// ---------------------------------------------------------------------------
// VQ argmin: dist = ||e||^2 - 2 z.e via WMMA.  4 waves x 16 vectors per block.
// ---------------------------------------------------------------------------
__global__ __launch_bounds__(128)
void vq_argmin_kernel(const float* __restrict__ z, const float* __restrict__ E,
                      const float* __restrict__ enorm, int* __restrict__ idx)
{
    __shared__ bf16_t Az[64][72];
    __shared__ bf16_t Bz[16][72];
    __shared__ float  Dist[4][16][17];

    const int tid  = threadIdx.x;
    const int lane = tid & 31;
    const int wid  = tid >> 5;
    const int r16  = lane & 15;
    const int lh   = lane >> 4;
    const int mbase = (int)blockIdx.x * 64;

    // stage 64 z-vectors: thread owns (row m, half of k); batched 8 loads -> b128
    {
        int m   = tid & 63;
        int kh0 = (tid >> 6) * 32;
        int gm  = mbase + m;
        int nb  = gm >> 12;
        int rem = gm & 4095;
        const float* zp = z + (((nb << 6) + kh0) << 12) + rem;
        for (int k0 = 0; k0 < 32; k0 += 8) {
            float tmp[8];
#pragma unroll
            for (int u = 0; u < 8; ++u) tmp[u] = zp[(k0 + u) << 12];
            v8bf pk;
#pragma unroll
            for (int u = 0; u < 8; ++u) pk[u] = (bf16_t)tmp[u];
            *(v8bf*)&Az[m][kh0 + k0] = pk;
        }
    }

    float best = 3.4e38f;
    int bestn = 0;

    for (int ct = 0; ct < 32; ++ct) {
        { // stage 16-code tile: contiguous f32 row chunk -> bf16 b128 store
            int n  = tid >> 3;
            int k0 = (tid & 7) * 8;
            const float* Ep = E + (((ct << 4) + n) << 6) + k0;
            float tmp[8];
#pragma unroll
            for (int u = 0; u < 8; ++u) tmp[u] = Ep[u];
            v8bf pk;
#pragma unroll
            for (int u = 0; u < 8; ++u) pk[u] = (bf16_t)tmp[u];
            *(v8bf*)&Bz[n][k0] = pk;
        }
        __syncthreads();

        v8f acc = {};
#pragma unroll
        for (int ks = 0; ks < 2; ++ks) {
            v8bf blo = *(const v8bf*)&Bz[r16][ks * 32 + lh * 8];
            v8bf bhi = *(const v8bf*)&Bz[r16][ks * 32 + 16 + lh * 8];
            v16bf bfr;
#pragma unroll
            for (int q = 0; q < 8; ++q) { bfr[q] = blo[q]; bfr[8 + q] = bhi[q]; }
            v8bf alo = *(const v8bf*)&Az[wid * 16 + r16][ks * 32 + lh * 8];
            v8bf ahi = *(const v8bf*)&Az[wid * 16 + r16][ks * 32 + 16 + lh * 8];
            v16bf afr;
#pragma unroll
            for (int q = 0; q < 8; ++q) { afr[q] = alo[q]; afr[8 + q] = ahi[q]; }
            acc = __builtin_amdgcn_wmma_f32_16x16x32_bf16(
                false, afr, false, bfr, (short)0, acc, false, false);
        }
#pragma unroll
        for (int r = 0; r < 8; ++r)
            Dist[wid][lh * 8 + r][r16] = enorm[(ct << 4) + r16] - 2.0f * acc[r];
        __syncthreads();

        if (lane < 16) {
            for (int c = 0; c < 16; ++c) {
                float dv = Dist[wid][lane][c];
                if (dv < best) { best = dv; bestn = (ct << 4) + c; }
            }
        }
        __syncthreads();
    }
    if (lane < 16) idx[mbase + wid * 16 + lane] = bestn;
}

__global__ void enorm_kernel(const float* __restrict__ E, float* __restrict__ enorm)
{
    int i = blockIdx.x * blockDim.x + threadIdx.x;
    if (i < 512) {
        float s = 0.0f;
        for (int k = 0; k < 64; ++k) { float v = E[(i << 6) + k]; s += v * v; }
        enorm[i] = s;
    }
}

__global__ void vq_init_kernel(float* __restrict__ counts, float* __restrict__ losssum)
{
    int i = blockIdx.x * blockDim.x + threadIdx.x;
    if (i < 512) counts[i] = 0.0f;
    if (i == 0) losssum[0] = 0.0f;
}

__global__ __launch_bounds__(256)
void vq_gather_kernel(const float* __restrict__ z, const float* __restrict__ E,
                      const int* __restrict__ idx, float* __restrict__ zq,
                      float* __restrict__ losssum)
{
    __shared__ float red[256];
    int t = blockIdx.x * blockDim.x + threadIdx.x;     // over 131072*64
    int p = t >> 6, k = t & 63;
    int nb = p >> 12, rem = p & 4095;
    int off = (((nb << 6) + k) << 12) + rem;
    float zv = z[off];
    float qv = E[(idx[p] << 6) + k];
    zq[off] = qv;
    float d = qv - zv;
    red[threadIdx.x] = d * d;
    __syncthreads();
    for (int s = 128; s > 0; s >>= 1) {
        if ((int)threadIdx.x < s) red[threadIdx.x] += red[threadIdx.x + s];
        __syncthreads();
    }
    if (threadIdx.x == 0) atomicAdd(losssum, red[0]);
}

__global__ __launch_bounds__(256)
void vq_hist_kernel(const int* __restrict__ idx, float* __restrict__ counts, int Mvec)
{
    __shared__ float h[512];
    for (int i = threadIdx.x; i < 512; i += blockDim.x) h[i] = 0.0f;
    __syncthreads();
    int t = blockIdx.x * blockDim.x + threadIdx.x;
    if (t < Mvec) atomicAdd(&h[idx[t]], 1.0f);
    __syncthreads();
    for (int i = threadIdx.x; i < 512; i += blockDim.x)
        if (h[i] != 0.0f) atomicAdd(&counts[i], h[i]);
}

__global__ __launch_bounds__(512)
void vq_finalize_kernel(const float* __restrict__ losssum,
                        const float* __restrict__ counts,
                        float* __restrict__ out_loss, float* __restrict__ out_perp,
                        float inv_nelem, float inv_mvec)
{
    __shared__ float red[512];
    int t = threadIdx.x;
    float p = counts[t] * inv_mvec;
    red[t] = p * logf(p + 1e-10f);
    __syncthreads();
    for (int s = 256; s > 0; s >>= 1) {
        if (t < s) red[t] += red[t + s];
        __syncthreads();
    }
    if (t == 0) {
        *out_perp = expf(-red[0]);
        *out_loss = (1.0f + BETA_CONST) * losssum[0] * inv_nelem;
    }
}

// ---------------------------------------------------------------------------
// Host-side orchestration
// ---------------------------------------------------------------------------
static inline int ilog2i(int v) { int l = 0; while ((1 << l) < v) ++l; return l; }

template <typename TIN, typename TOUT, int WM, int WN>
static void run_conv(const TIN* in, const bf16_t* wT, const float* bias,
                     const TOUT* resid, TOUT* out,
                     int Nb, int Cin, int Hin, int Win,
                     int Cout, int Hout, int Wout,
                     int KH, int KW, int stride, int pad,
                     int in_relu, int out_relu, int transposed, hipStream_t s)
{
    const int lWin    = ilog2i(Win);
    const int lHinWin = ilog2i(Hin * Win);
    const int lWout   = ilog2i(Wout);
    const int lHoW    = ilog2i(Hout * Wout);
    const int sshift  = (stride == 2) ? 1 : 0;
    const int M = Nb * Hout * Wout;
    const int K = Cin * KH * KW;
    const int Kpad = ((K + 63) / 64) * 64;
    const int BLOCK_M = WM * 64, BLOCK_N = WN * 16;
    const int Mtiles = M / BLOCK_M;
    const int Ntiles = (Cout + BLOCK_N - 1) / BLOCK_N;
    conv_wmma<TIN, TOUT, WM, WN><<<Mtiles * Ntiles, 128, 0, s>>>(
        in, wT, bias, resid, out,
        Cin, Hin, Win, lHinWin, lWin,
        Cout, lHoW, lWout, Wout,
        KH, KW, K, Kpad, sshift, pad, M, in_relu, out_relu, transposed);
}

extern "C" void kernel_launch(void* const* d_in, const int* in_sizes, int n_in,
                              void* d_out, int out_size, void* d_ws, size_t ws_size,
                              hipStream_t stream)
{
    (void)in_sizes; (void)n_in; (void)ws_size; (void)out_size;
    const float* x    = (const float*)d_in[0];
    const float* w1   = (const float*)d_in[1];
    const float* b1   = (const float*)d_in[2];
    const float* w2   = (const float*)d_in[3];
    const float* b2   = (const float*)d_in[4];
    const float* w3   = (const float*)d_in[5];
    const float* b3   = (const float*)d_in[6];
    const float* r0a  = (const float*)d_in[7];
    const float* r0b  = (const float*)d_in[8];
    const float* r1a  = (const float*)d_in[9];
    const float* r1b  = (const float*)d_in[10];
    const float* wp   = (const float*)d_in[11];
    const float* bp   = (const float*)d_in[12];
    const float* E    = (const float*)d_in[13];
    const float* dw1  = (const float*)d_in[14];
    const float* db1  = (const float*)d_in[15];
    const float* dr0a = (const float*)d_in[16];
    const float* dr0b = (const float*)d_in[17];
    const float* dr1a = (const float*)d_in[18];
    const float* dr1b = (const float*)d_in[19];
    const float* tw1  = (const float*)d_in[20];
    const float* tb1  = (const float*)d_in[21];
    const float* tw2  = (const float*)d_in[22];
    const float* tb2  = (const float*)d_in[23];

    float* out = (float*)d_out;

    // --------- workspace carve-up ---------
    bf16_t* zA = (bf16_t*)d_ws;            // 33,554,432 bf16 (conv1 / tconv1 out)
    bf16_t* zB = zA + 33554432;            // 16,777,216 bf16
    bf16_t* zC = zB + 16777216;            // 16,777,216 bf16
    bf16_t* zE = zC + 16777216;            //  4,194,304 bf16 (res hidden)
    float*  zD = (float*)(zE + 4194304);   //  8,388,608 f32  (pre-VQ z)
    float*  zQ = zD + 8388608;             //  8,388,608 f32  (quantized)
    float*  enorm   = zQ + 8388608;        // 512
    float*  counts  = enorm + 512;         // 512
    float*  losssum = counts + 512;        // 1 (+31 pad for alignment)
    int*    idxbuf  = (int*)(losssum + 32);// 131,072
    bf16_t* wtp = (bf16_t*)(idxbuf + 131072);  // 16B aligned

    // --------- prep all weights to bf16 [COUTP][Kpad] ---------
    struct WSpec { const float* w; int Cout, Cin, KH, KW, tr; };
    const WSpec specs[15] = {
        { w1,   64,   3, 4, 4, 0 }, { w2,  128,  64, 4, 4, 0 }, { w3, 128, 128, 3, 3, 0 },
        { r0a,  32, 128, 3, 3, 0 }, { r0b, 128,  32, 3, 3, 0 },
        { r1a,  32, 128, 3, 3, 0 }, { r1b, 128,  32, 3, 3, 0 },
        { wp,   64, 128, 1, 1, 0 },
        { dw1, 128,  64, 3, 3, 0 },
        { dr0a, 32, 128, 3, 3, 0 }, { dr0b, 128, 32, 3, 3, 0 },
        { dr1a, 32, 128, 3, 3, 0 }, { dr1b, 128, 32, 3, 3, 0 },
        { tw1,  64, 128, 4, 4, 1 }, { tw2,   3,  64, 4, 4, 1 },
    };
    bf16_t* wt[15];
    {
        bf16_t* p = wtp;
        for (int i = 0; i < 15; ++i) {
            wt[i] = p;
            int K = specs[i].Cin * specs[i].KH * specs[i].KW;
            int Kpad = ((K + 63) / 64) * 64;
            int total = COUTP * Kpad;
            prep_weight_kernel<<<(total + 255) / 256, 256, 0, stream>>>(
                specs[i].w, wt[i], specs[i].Cout, specs[i].Cin,
                specs[i].KH, specs[i].KW, Kpad, specs[i].tr);
            p += total;
        }
    }

    // --------- encoder ---------
    run_conv<float,  bf16_t, 1, 4>(x,  wt[0], b1, (bf16_t*)nullptr, zA, 32,   3, 256, 256,  64, 128, 128, 4, 4, 2, 1, 0, 1, 0, stream);
    run_conv<bf16_t, bf16_t, 1, 4>(zA, wt[1], b2, (bf16_t*)nullptr, zB, 32,  64, 128, 128, 128,  64,  64, 4, 4, 2, 1, 0, 1, 0, stream);
    run_conv<bf16_t, bf16_t, 1, 4>(zB, wt[2], b3, (bf16_t*)nullptr, zC, 32, 128,  64,  64, 128,  64,  64, 3, 3, 1, 1, 0, 0, 0, stream);
    // residual pair 0
    run_conv<bf16_t, bf16_t, 2, 2>(zC, wt[3], nullptr, (bf16_t*)nullptr, zE, 32, 128, 64, 64,  32, 64, 64, 3, 3, 1, 1, 1, 1, 0, stream);
    run_conv<bf16_t, bf16_t, 1, 4>(zE, wt[4], nullptr, zC,               zC, 32,  32, 64, 64, 128, 64, 64, 3, 3, 1, 1, 0, 0, 0, stream);
    // residual pair 1
    run_conv<bf16_t, bf16_t, 2, 2>(zC, wt[5], nullptr, (bf16_t*)nullptr, zE, 32, 128, 64, 64,  32, 64, 64, 3, 3, 1, 1, 1, 1, 0, stream);
    run_conv<bf16_t, bf16_t, 1, 4>(zE, wt[6], nullptr, zC,               zC, 32,  32, 64, 64, 128, 64, 64, 3, 3, 1, 1, 0, 0, 0, stream);
    // pre-VQ 1x1 (input relu = res-stack final relu); fp32 out for VQ numerics
    run_conv<bf16_t, float, 1, 4>(zC, wt[7], bp, (float*)nullptr, zD, 32, 128, 64, 64, 64, 64, 64, 1, 1, 1, 0, 1, 0, 0, stream);

    // --------- vector quantization ---------
    const int Mvec = 32 * 64 * 64;
    enorm_kernel<<<2, 256, 0, stream>>>(E, enorm);
    vq_init_kernel<<<2, 256, 0, stream>>>(counts, losssum);
    vq_argmin_kernel<<<Mvec / 64, 128, 0, stream>>>(zD, E, enorm, idxbuf);
    vq_gather_kernel<<<(Mvec * 64) / 256, 256, 0, stream>>>(zD, E, idxbuf, zQ, losssum);
    vq_hist_kernel<<<(Mvec + 255) / 256, 256, 0, stream>>>(idxbuf, counts, Mvec);

    // --------- decoder ---------
    run_conv<float,  bf16_t, 1, 4>(zQ, wt[8], db1, (bf16_t*)nullptr, zB, 32, 64, 64, 64, 128, 64, 64, 3, 3, 1, 1, 0, 0, 0, stream);
    run_conv<bf16_t, bf16_t, 2, 2>(zB, wt[9],  nullptr, (bf16_t*)nullptr, zE, 32, 128, 64, 64,  32, 64, 64, 3, 3, 1, 1, 1, 1, 0, stream);
    run_conv<bf16_t, bf16_t, 1, 4>(zE, wt[10], nullptr, zB,               zB, 32,  32, 64, 64, 128, 64, 64, 3, 3, 1, 1, 0, 0, 0, stream);
    run_conv<bf16_t, bf16_t, 2, 2>(zB, wt[11], nullptr, (bf16_t*)nullptr, zE, 32, 128, 64, 64,  32, 64, 64, 3, 3, 1, 1, 1, 1, 0, stream);
    run_conv<bf16_t, bf16_t, 1, 4>(zE, wt[12], nullptr, zB,               zB, 32,  32, 64, 64, 128, 64, 64, 3, 3, 1, 1, 0, 0, 0, stream);
    // transposed convs (gather form)
    run_conv<bf16_t, bf16_t, 1, 4>(zB, wt[13], tb1, (bf16_t*)nullptr, zA,      32, 128,  64,  64, 64, 128, 128, 4, 4, 2, 1, 1, 1, 1, stream);
    run_conv<bf16_t, float,  4, 1>(zA, wt[14], tb2, (float*)nullptr,  out + 1, 32,  64, 128, 128,  3, 256, 256, 4, 4, 2, 1, 0, 0, 1, stream);

    // --------- scalars ---------
    vq_finalize_kernel<<<1, 512, 0, stream>>>(losssum, counts,
                                              out, out + 1 + 32 * 3 * 256 * 256,
                                              1.0f / (float)(Mvec * 64),
                                              1.0f / (float)Mvec);
}
